// TopKastLinear_39487929319525
// MI455X (gfx1250) — matrix operand order repaired and checked by reading, not compiled, not run on hardware
//
#include <hip/hip_runtime.h>
#include <hip/hip_bf16.h>

typedef __attribute__((ext_vector_type(16))) __bf16 v16bf;
typedef __attribute__((ext_vector_type(8)))  __bf16 v8bf;
typedef __attribute__((ext_vector_type(4)))  __bf16 v4bf;
typedef __attribute__((ext_vector_type(8)))  float  v8f;

#define MDIM 8192
#define NDIM 4096
#define KDIM 4096
#define NW   (NDIM * KDIM)          // 16777216 weights
#define K0_RANK 15938354u           // floor(0.95 * (NW-1)); frac = 0.25

// ws layout (uint32 words):
// [0] prefix  [1] remaining  [2..257] bins  [258] count_le  [259] min_greater  [260] thr bits

// ---------------------------------------------------------------------------
// Radix select (exact k-th order statistic of |w| bit patterns)
// ---------------------------------------------------------------------------
__global__ void sel_init(unsigned* ws) {
    const int t = threadIdx.x;
    ws[2 + t] = 0u;
    if (t == 0) {
        ws[0]   = 0u;          // prefix
        ws[1]   = K0_RANK;     // remaining rank (0-indexed, ascending)
        ws[258] = 0u;          // count_le
        ws[259] = 0xFFFFFFFFu; // min_greater
        ws[260] = 0u;          // thr
    }
}

__global__ void sel_hist(const float* __restrict__ w, unsigned* __restrict__ ws,
                         int shift, unsigned mask) {
    __shared__ unsigned lbins[256];
    const int tid = threadIdx.x;
    lbins[tid] = 0u;
    __syncthreads();

    const unsigned prefix = ws[0];
    const float4* w4 = (const float4*)w;
    const int n4 = NW / 4;
    for (int i = blockIdx.x * blockDim.x + tid; i < n4; i += gridDim.x * blockDim.x) {
        float4 v = w4[i];
        unsigned b0 = __float_as_uint(v.x) & 0x7FFFFFFFu;
        unsigned b1 = __float_as_uint(v.y) & 0x7FFFFFFFu;
        unsigned b2 = __float_as_uint(v.z) & 0x7FFFFFFFu;
        unsigned b3 = __float_as_uint(v.w) & 0x7FFFFFFFu;
        if ((b0 & mask) == prefix) atomicAdd(&lbins[(b0 >> shift) & 0xFFu], 1u);
        if ((b1 & mask) == prefix) atomicAdd(&lbins[(b1 >> shift) & 0xFFu], 1u);
        if ((b2 & mask) == prefix) atomicAdd(&lbins[(b2 >> shift) & 0xFFu], 1u);
        if ((b3 & mask) == prefix) atomicAdd(&lbins[(b3 >> shift) & 0xFFu], 1u);
    }
    __syncthreads();
    const unsigned c = lbins[tid];
    if (c) atomicAdd(&ws[2 + tid], c);
}

__global__ void sel_pick(unsigned* ws, int shift) {
    if (threadIdx.x == 0) {
        unsigned r = ws[1];
        unsigned cum = 0u;
        int b = 0;
        for (; b < 256; ++b) {
            unsigned c = ws[2 + b];
            if (cum + c > r) break;
            cum += c;
        }
        ws[0] |= ((unsigned)b) << shift;
        ws[1] = r - cum;
    }
    __syncthreads();
    ws[2 + threadIdx.x] = 0u;  // reset bins for next pass
}

__global__ void sel_count(const float* __restrict__ w, unsigned* __restrict__ ws) {
    const unsigned pv = ws[0];
    unsigned cnt = 0u, mn = 0xFFFFFFFFu;
    const float4* w4 = (const float4*)w;
    const int n4 = NW / 4;
    for (int i = blockIdx.x * blockDim.x + threadIdx.x; i < n4; i += gridDim.x * blockDim.x) {
        float4 v = w4[i];
        unsigned b[4] = { __float_as_uint(v.x) & 0x7FFFFFFFu, __float_as_uint(v.y) & 0x7FFFFFFFu,
                          __float_as_uint(v.z) & 0x7FFFFFFFu, __float_as_uint(v.w) & 0x7FFFFFFFu };
        #pragma unroll
        for (int j = 0; j < 4; ++j) {
            if (b[j] <= pv) ++cnt;
            else if (b[j] < mn) mn = b[j];
        }
    }
    if (cnt) atomicAdd(&ws[258], cnt);
    atomicMin(&ws[259], mn);
}

__global__ void sel_final(unsigned* ws) {
    float v0 = __uint_as_float(ws[0]);
    // a[k0+1] == v0 if at least k0+2 elements are <= v0, else next greater value
    float v1 = (ws[258] >= K0_RANK + 2u) ? v0 : __uint_as_float(ws[259]);
    float thr = v0 + 0.25f * (v1 - v0);   // linear interp at frac 0.25
    ws[260] = __float_as_uint(thr);
}

// ---------------------------------------------------------------------------
// Masked GEMM: Y = X @ (mask(W))^T + bias   (bf16 WMMA, fp32 accumulate)
// Block tile 128x256, BK=32, 8 waves (2x4), wave tile 64x64 = 4x4 WMMA tiles.
// Ping-pong double-buffered LDS, one barrier per k-step, global loads for
// tile k+2 issued under the compute of tile k.
// ---------------------------------------------------------------------------
#define BM 128
#define BN 256
#define BK 32
#define NK (KDIM / BK)
#define LDSP 40   // padded LDS row stride in bf16 (80B; 5 odd => conflict-free b128)

__global__ __launch_bounds__(256)
void gemm_topkast(const float* __restrict__ X, const float* __restrict__ W,
                  const float* __restrict__ bias, float* __restrict__ Y,
                  const unsigned* __restrict__ ws) {
    __shared__ __align__(16) __bf16 sA[2][BM * LDSP];   // 20 KB
    __shared__ __align__(16) __bf16 sB[2][BN * LDSP];   // 40 KB

    const float thr = __uint_as_float(ws[260]);

    const int tid  = threadIdx.x;
    const int lane = tid & 31;
    const int wave = tid >> 5;       // 0..7
    const int lr   = lane & 15;
    const int lh   = lane >> 4;

    const int n0 = blockIdx.x * BN;
    const int m0 = blockIdx.y * BM;
    const int wm = (wave >> 2) * 64; // wave M offset: 0 or 64
    const int wn = (wave & 3) * 64;  // wave N offset: 0,64,128,192

    // global tile loaders: A = 1024 float4 (4/thread), B = 2048 float4 (8/thread)
    const int fr = tid >> 3;         // row group base (0..31)
    const int fc = (tid & 7) * 4;    // column in floats (0..28)

    const float* Ab = X + (size_t)(m0 + fr) * KDIM + fc;
    const float* Bb = W + (size_t)(n0 + fr) * KDIM + fc;

    v8f acc[4][4];
    #pragma unroll
    for (int i = 0; i < 4; ++i)
        #pragma unroll
        for (int j = 0; j < 4; ++j) {
            v8f z = {0.f, 0.f, 0.f, 0.f, 0.f, 0.f, 0.f, 0.f};
            acc[i][j] = z;
        }

    float4 ar[4];
    float4 br[8];

    // ---- prologue: tile 0 -> buf 0, then issue loads for tile 1 ----
    #pragma unroll
    for (int i = 0; i < 4; ++i) ar[i] = *(const float4*)(Ab + (size_t)i * 32 * KDIM);
    #pragma unroll
    for (int i = 0; i < 8; ++i) br[i] = *(const float4*)(Bb + (size_t)i * 32 * KDIM);

    #pragma unroll
    for (int i = 0; i < 4; ++i) {
        v4bf b;
        b[0] = (__bf16)ar[i].x; b[1] = (__bf16)ar[i].y;
        b[2] = (__bf16)ar[i].z; b[3] = (__bf16)ar[i].w;
        *(v4bf*)(&sA[0][(fr + i * 32) * LDSP + fc]) = b;
    }
    #pragma unroll
    for (int i = 0; i < 8; ++i) {
        float4 v = br[i];
        v.x = (fabsf(v.x) >= thr) ? v.x : 0.0f;
        v.y = (fabsf(v.y) >= thr) ? v.y : 0.0f;
        v.z = (fabsf(v.z) >= thr) ? v.z : 0.0f;
        v.w = (fabsf(v.w) >= thr) ? v.w : 0.0f;
        v4bf b;
        b[0] = (__bf16)v.x; b[1] = (__bf16)v.y; b[2] = (__bf16)v.z; b[3] = (__bf16)v.w;
        *(v4bf*)(&sB[0][(fr + i * 32) * LDSP + fc]) = b;
    }

    #pragma unroll
    for (int i = 0; i < 4; ++i) ar[i] = *(const float4*)(Ab + (size_t)i * 32 * KDIM + BK);
    #pragma unroll
    for (int i = 0; i < 8; ++i) br[i] = *(const float4*)(Bb + (size_t)i * 32 * KDIM + BK);

    __syncthreads();

    // ---- main loop: one barrier per k-step ----
    for (int kt = 0; kt < NK; ++kt) {
        const int p = kt & 1;

        // B fragments: lane lr = N col; contiguous K chunk at lh*16
        v16bf bfr[4];
        #pragma unroll
        for (int t = 0; t < 4; ++t) {
            const __bf16* pB = &sB[p][(wn + t * 16 + lr) * LDSP + lh * 16];
            const v8bf lo = *(const v8bf*)(pB);
            const v8bf hi = *(const v8bf*)(pB + 8);
            #pragma unroll
            for (int e = 0; e < 8; ++e) { bfr[t][e] = lo[e]; bfr[t][e + 8] = hi[e]; }
        }
        // A fragment per tm, reused across 4 N tiles
        #pragma unroll
        for (int tm = 0; tm < 4; ++tm) {
            const __bf16* pA = &sA[p][(wm + tm * 16 + lr) * LDSP];
            const v8bf lo = *(const v8bf*)(pA + lh * 16);
            const v8bf hi = *(const v8bf*)(pA + lh * 16 + 8);
            v16bf a;
            #pragma unroll
            for (int e = 0; e < 8; ++e) { a[e] = lo[e]; a[e + 8] = hi[e]; }
            #pragma unroll
            for (int tn = 0; tn < 4; ++tn)
                acc[tm][tn] = __builtin_amdgcn_wmma_f32_16x16x32_bf16(
                    false, a, false, bfr[tn], (short)0, acc[tm][tn], false, false);
        }

        if (kt + 1 < NK) {
            // stage tile kt+1 (already in regs) into the other buffer
            #pragma unroll
            for (int i = 0; i < 4; ++i) {
                v4bf b;
                b[0] = (__bf16)ar[i].x; b[1] = (__bf16)ar[i].y;
                b[2] = (__bf16)ar[i].z; b[3] = (__bf16)ar[i].w;
                *(v4bf*)(&sA[1 - p][(fr + i * 32) * LDSP + fc]) = b;
            }
            #pragma unroll
            for (int i = 0; i < 8; ++i) {
                float4 v = br[i];
                v.x = (fabsf(v.x) >= thr) ? v.x : 0.0f;
                v.y = (fabsf(v.y) >= thr) ? v.y : 0.0f;
                v.z = (fabsf(v.z) >= thr) ? v.z : 0.0f;
                v.w = (fabsf(v.w) >= thr) ? v.w : 0.0f;
                v4bf b;
                b[0] = (__bf16)v.x; b[1] = (__bf16)v.y;
                b[2] = (__bf16)v.z; b[3] = (__bf16)v.w;
                *(v4bf*)(&sB[1 - p][(fr + i * 32) * LDSP + fc]) = b;
            }
            if (kt + 2 < NK) {
                const int k2 = (kt + 2) * BK;
                #pragma unroll
                for (int i = 0; i < 4; ++i)
                    ar[i] = *(const float4*)(Ab + (size_t)i * 32 * KDIM + k2);
                #pragma unroll
                for (int i = 0; i < 8; ++i)
                    br[i] = *(const float4*)(Bb + (size_t)i * 32 * KDIM + k2);
            }
        }
        __syncthreads();
    }

    // epilogue: C layout -> row = base + e + lh*8, col = lr ; add bias, store fp32
    #pragma unroll
    for (int tn = 0; tn < 4; ++tn) {
        const int col = n0 + wn + tn * 16 + lr;
        const float bv = bias[col];
        #pragma unroll
        for (int tm = 0; tm < 4; ++tm) {
            const int rbase = m0 + wm + tm * 16 + lh * 8;
            #pragma unroll
            for (int e = 0; e < 8; ++e)
                Y[(size_t)(rbase + e) * NDIM + col] = acc[tm][tn][e] + bv;
        }
    }
}

// ---------------------------------------------------------------------------
extern "C" void kernel_launch(void* const* d_in, const int* in_sizes, int n_in,
                              void* d_out, int out_size, void* d_ws, size_t ws_size,
                              hipStream_t stream) {
    const float* X    = (const float*)d_in[0];
    const float* W    = (const float*)d_in[1];
    const float* bias = (const float*)d_in[2];
    float* Y          = (float*)d_out;
    unsigned* ws      = (unsigned*)d_ws;

    sel_init<<<1, 256, 0, stream>>>(ws);
    for (int pass = 0; pass < 4; ++pass) {
        const int shift = 24 - 8 * pass;
        const unsigned mask = (pass == 0) ? 0u : (0xFFFFFFFFu << (shift + 8));
        sel_hist<<<2048, 256, 0, stream>>>(W, ws, shift, mask);
        sel_pick<<<1, 256, 0, stream>>>(ws, shift);
    }
    sel_count<<<2048, 256, 0, stream>>>(W, ws);
    sel_final<<<1, 1, 0, stream>>>(ws);

    dim3 grid(NDIM / BN, MDIM / BM);
    gemm_topkast<<<grid, 256, 0, stream>>>(X, W, bias, Y, ws);
}
